// Model_62242666053960
// MI455X (gfx1250) — matrix operand order repaired and checked by reading, not compiled, not run on hardware
//
#include <hip/hip_runtime.h>

typedef __attribute__((ext_vector_type(8)))  float  v8f;
typedef __attribute__((ext_vector_type(16))) __bf16 v16bf;

__device__ inline unsigned short f32_to_bf16(float f) {
  unsigned int u = __float_as_uint(f);
  u += 0x7FFFu + ((u >> 16) & 1u);          // round-to-nearest-even
  return (unsigned short)(u >> 16);
}

// ---------------------------------------------------------------------------
// Core: masked gather-GEMM over 27 offsets using v_wmma_f32_16x16x32_bf16.
// One wave32 computes a 32(sites) x 32(out-ch) tile: 2 A fragments x 2 B
// fragments -> 4 WMMAs per fetch group (2x operand reuse vs 16x16 tiles).
// KCH = Cin/32 is a template parameter so the K loop fully unrolls and the
// scheduler can pipeline gathers against the WMMA groups.
//
// A layout (ISA 16-bit A 16x32): lane L -> row M=L&15, halves [hi*8,+8) and
// [16+hi*8,+8) of each 32-half K chunk. B layout (ISA): lane L -> column
// N=L&15, halves [hi*8,+8) and [16+hi*8,+8) of W^T row (weights packed
// [k][Cout][Cin]). EXEC stays all-ones; masking is done by zeroing A data.
// ---------------------------------------------------------------------------
template <int KCH>
__global__ __launch_bounds__(128) void subconv_wmma_kernel(
    const unsigned short* __restrict__ xb,   // [N][Cin] bf16
    const unsigned short* __restrict__ wb,   // [27][Cout][Cin] bf16 (pre-transposed)
    const int* __restrict__ nbr,             // [N][27]
    const unsigned char* __restrict__ mask,  // [N][27]
    float* __restrict__ out,                 // [N][Cout]
    int N, int Cout) {
  constexpr int Cin = KCH * 32;
  const int tilesM = (N + 31) >> 5;
  const int tm = blockIdx.x * blockDim.y + threadIdx.y;
  if (tm >= tilesM) return;                  // wave-uniform exit
  const int n0   = blockIdx.y << 5;
  const int lane = threadIdx.x;
  const int mrow = lane & 15;
  const int hi   = lane >> 4;
  const int m0   = tm << 5;

  const int  site0 = m0 + mrow;              // rows 0..15 of the tile
  const int  site1 = m0 + 16 + mrow;         // rows 16..31
  const bool v0 = site0 < N;
  const bool v1 = site1 < N;
  const int  s0 = v0 ? site0 : 0;
  const int  s1 = v1 ? site1 : 0;

  v8f acc00 = {}, acc01 = {}, acc10 = {}, acc11 = {};

  for (int k = 0; k < 27; ++k) {
    const int  nb0 = nbr[s0 * 27 + k];
    const int  nb1 = nbr[s1 * 27 + k];
    const bool a0  = v0 && (mask[s0 * 27 + k] != 0);
    const bool a1  = v1 && (mask[s1 * 27 + k] != 0);
    if (k + 1 < 27) {                        // pull next offset's gather rows in
      __builtin_prefetch(xb + (size_t)nbr[s0 * 27 + k + 1] * Cin, 0, 1);
      __builtin_prefetch(xb + (size_t)nbr[s1 * 27 + k + 1] * Cin, 0, 1);
    }
    const uint4* ap0 = (const uint4*)(xb + (size_t)nb0 * Cin + hi * 8);
    const uint4* ap1 = (const uint4*)(xb + (size_t)nb1 * Cin + hi * 8);
    const uint4* bp0 = (const uint4*)(wb + ((size_t)k * Cout + n0 + mrow) * Cin + hi * 8);
    const uint4* bp1 = (const uint4*)(wb + ((size_t)k * Cout + n0 + 16 + mrow) * Cin + hi * 8);
#pragma unroll
    for (int kc = 0; kc < KCH; ++kc) {
      union { v16bf v; uint4 u[2]; } A0, A1, B0, B1;
      if (a0) { A0.u[0] = ap0[0]; A0.u[1] = ap0[2]; }
      else    { A0.u[0] = make_uint4(0,0,0,0); A0.u[1] = make_uint4(0,0,0,0); }
      if (a1) { A1.u[0] = ap1[0]; A1.u[1] = ap1[2]; }
      else    { A1.u[0] = make_uint4(0,0,0,0); A1.u[1] = make_uint4(0,0,0,0); }
      B0.u[0] = bp0[0]; B0.u[1] = bp0[2];
      B1.u[0] = bp1[0]; B1.u[1] = bp1[2];
      acc00 = __builtin_amdgcn_wmma_f32_16x16x32_bf16(false, A0.v, false, B0.v, (short)0, acc00, false, false);
      acc01 = __builtin_amdgcn_wmma_f32_16x16x32_bf16(false, A0.v, false, B1.v, (short)0, acc01, false, false);
      acc10 = __builtin_amdgcn_wmma_f32_16x16x32_bf16(false, A1.v, false, B0.v, (short)0, acc10, false, false);
      acc11 = __builtin_amdgcn_wmma_f32_16x16x32_bf16(false, A1.v, false, B1.v, (short)0, acc11, false, false);
      ap0 += 4; ap1 += 4; bp0 += 4; bp1 += 4;        // advance 32 halves
    }
  }
  // C/D layout: VGPR r -> M = r + 8*hi, N = lane&15
#pragma unroll
  for (int r = 0; r < 8; ++r) {
    int ma = m0 + hi * 8 + r;                // m-half 0
    int mb = m0 + 16 + hi * 8 + r;           // m-half 1
    if (ma < N) {
      out[(size_t)ma * Cout + n0 + mrow]      = acc00[r];
      out[(size_t)ma * Cout + n0 + 16 + mrow] = acc01[r];
    }
    if (mb < N) {
      out[(size_t)mb * Cout + n0 + mrow]      = acc10[r];
      out[(size_t)mb * Cout + n0 + 16 + mrow] = acc11[r];
    }
  }
}

// ---------------------------------------------------------------------------
// BatchNorm: pass 1 computes per-channel mean & rsqrt(var+eps)
// ---------------------------------------------------------------------------
__global__ __launch_bounds__(256) void bn_stats_kernel(
    const float* __restrict__ x, int N, int C, float* __restrict__ stats) {
  __shared__ float s1[256], s2[256];
  const int c = blockIdx.x;
  float a = 0.f, b = 0.f;
  for (int i = threadIdx.x; i < N; i += 256) {
    float v = x[(size_t)i * C + c];
    a += v; b += v * v;
  }
  s1[threadIdx.x] = a; s2[threadIdx.x] = b;
  __syncthreads();
  for (int off = 128; off > 0; off >>= 1) {
    if (threadIdx.x < off) {
      s1[threadIdx.x] += s1[threadIdx.x + off];
      s2[threadIdx.x] += s2[threadIdx.x + off];
    }
    __syncthreads();
  }
  if (threadIdx.x == 0) {
    float mu  = s1[0] / (float)N;
    float var = s2[0] / (float)N - mu * mu;
    stats[c]     = mu;
    stats[C + c] = rsqrtf(var + 1e-4f);
  }
}

// pass 2: normalize + ReLU; optionally emit f32 and/or bf16 copies
__global__ __launch_bounds__(256) void bn_apply_kernel(
    const float* __restrict__ x, const float* __restrict__ stats,
    const float* __restrict__ g, const float* __restrict__ b,
    float* __restrict__ outf, unsigned short* __restrict__ outb,
    int N, int C) {
  size_t t = (size_t)blockIdx.x * 256 + threadIdx.x;
  if (t >= (size_t)N * C) return;
  int c = (int)(t % C);
  float v = (x[t] - stats[c]) * stats[C + c] * g[c] + b[c];
  v = v > 0.f ? v : 0.f;
  if (outf) outf[t] = v;
  if (outb) outb[t] = f32_to_bf16(v);
}

// pack f32 W[k][Cin][Cout] -> bf16 W^T[k][Cout][Cin]
__global__ __launch_bounds__(256) void wpack_kernel(
    const float* __restrict__ w, unsigned short* __restrict__ wb,
    int K, int Cin, int Cout) {
  size_t t = (size_t)blockIdx.x * 256 + threadIdx.x;
  if (t >= (size_t)K * Cin * Cout) return;
  int co = (int)(t % Cout);
  size_t r = t / Cout;
  int ci = (int)(r % Cin);
  int k  = (int)(r / Cin);
  wb[((size_t)k * Cout + co) * Cin + ci] = f32_to_bf16(w[t]);
}

// initial SubmanifoldConvolution(1 -> 32): tiny, scalar
__global__ __launch_bounds__(256) void subconv_in_kernel(
    const float* __restrict__ feats, const float* __restrict__ w,  // w[27][1][32]
    const int* __restrict__ nbr, const unsigned char* __restrict__ mask,
    float* __restrict__ out, int N) {
  size_t t = (size_t)blockIdx.x * 256 + threadIdx.x;
  if (t >= (size_t)N * 32) return;
  int c = (int)(t & 31);
  int i = (int)(t >> 5);
  float acc = 0.f;
  for (int k = 0; k < 27; ++k)
    if (mask[i * 27 + k]) acc += feats[nbr[i * 27 + k]] * w[k * 32 + c];
  out[t] = acc;
}

__global__ __launch_bounds__(256) void zero_kernel(float* __restrict__ p, size_t n) {
  size_t t = (size_t)blockIdx.x * 256 + threadIdx.x;
  if (t < n) p[t] = 0.f;
}

// strided 2x2x2 down-conv: per-site kernel offset kp[i], scatter-add to parent
__global__ __launch_bounds__(256) void downconv_kernel(
    const float* __restrict__ x, const int* __restrict__ par,
    const int* __restrict__ kp, const float* __restrict__ w,  // [8][Cin][Cout]
    float* __restrict__ outc, int N, int Cin, int Cout) {
  size_t t = (size_t)blockIdx.x * 256 + threadIdx.x;
  if (t >= (size_t)N * Cout) return;
  int oc = (int)(t % Cout);
  int i  = (int)(t / Cout);
  const float* wk = w + (size_t)kp[i] * Cin * Cout + oc;
  const float* xi = x + (size_t)i * Cin;
  float acc = 0.f;
  for (int c = 0; c < Cin; ++c) acc += xi[c] * wk[(size_t)c * Cout];
  atomicAdd(&outc[(size_t)par[i] * Cout + oc], acc);
}

// transposed conv: gather from parent, per-site kernel offset
__global__ __launch_bounds__(256) void deconv_kernel(
    const float* __restrict__ xc, const int* __restrict__ par,
    const int* __restrict__ kp, const float* __restrict__ w,  // [8][Cin][Cout]
    float* __restrict__ out, int N, int Cin, int Cout) {
  size_t t = (size_t)blockIdx.x * 256 + threadIdx.x;
  if (t >= (size_t)N * Cout) return;
  int oc = (int)(t % Cout);
  int i  = (int)(t / Cout);
  const float* wk = w + (size_t)kp[i] * Cin * Cout + oc;
  const float* xi = xc + (size_t)par[i] * Cin;
  float acc = 0.f;
  for (int c = 0; c < Cin; ++c) acc += xi[c] * wk[(size_t)c * Cout];
  out[t] = acc;
}

__global__ __launch_bounds__(256) void concat_kernel(
    const float* __restrict__ a, const float* __restrict__ b,
    float* __restrict__ cat, int N, int Ca, int Cb) {
  size_t t = (size_t)blockIdx.x * 256 + threadIdx.x;
  int C = Ca + Cb;
  if (t >= (size_t)N * C) return;
  int c = (int)(t % C);
  int i = (int)(t / C);
  cat[t] = (c < Ca) ? a[(size_t)i * Ca + c] : b[(size_t)i * Cb + (c - Ca)];
}

__global__ __launch_bounds__(256) void linear_kernel(
    const float* __restrict__ x, const float* __restrict__ w,  // [32][2]
    const float* __restrict__ bias, float* __restrict__ out, int N) {
  int i = blockIdx.x * 256 + threadIdx.x;
  if (i >= N) return;
  float a0 = bias[0], a1 = bias[1];
  const float* xi = x + (size_t)i * 32;
  for (int c = 0; c < 32; ++c) {
    float v = xi[c];
    a0 += v * w[c * 2 + 0];
    a1 += v * w[c * 2 + 1];
  }
  out[(size_t)i * 2 + 0] = a0;
  out[(size_t)i * 2 + 1] = a1;
}

// host-side dispatch over the (compile-time) K-chunk count
static void launch_subconv(const unsigned short* xb, const unsigned short* wb,
                           const int* nb, const unsigned char* mk, float* out,
                           int N, int Cin, int Cout, hipStream_t stream) {
  int tilesM = (N + 31) / 32;
  dim3 g((tilesM + 3) / 4, Cout / 32), blk(32, 4);
  switch (Cin / 32) {
    case 1: subconv_wmma_kernel<1><<<g, blk, 0, stream>>>(xb, wb, nb, mk, out, N, Cout); break;
    case 2: subconv_wmma_kernel<2><<<g, blk, 0, stream>>>(xb, wb, nb, mk, out, N, Cout); break;
    case 3: subconv_wmma_kernel<3><<<g, blk, 0, stream>>>(xb, wb, nb, mk, out, N, Cout); break;
    case 4: subconv_wmma_kernel<4><<<g, blk, 0, stream>>>(xb, wb, nb, mk, out, N, Cout); break;
    case 5: subconv_wmma_kernel<5><<<g, blk, 0, stream>>>(xb, wb, nb, mk, out, N, Cout); break;
    case 6: subconv_wmma_kernel<6><<<g, blk, 0, stream>>>(xb, wb, nb, mk, out, N, Cout); break;
    case 8: subconv_wmma_kernel<8><<<g, blk, 0, stream>>>(xb, wb, nb, mk, out, N, Cout); break;
    default: break;  // not reachable for this network
  }
}

// ---------------------------------------------------------------------------
// Host orchestration. Input leaves assume JAX pytree flattening order
// (dict keys sorted alphabetically, lists/tuples in order) -> 80 leaves:
//  0 feats | 1-10 bn1 | 11-18 bn2 | 19-26 bnd | 27-28 bnf | 29-36 bnu |
//  37-41 conv1 | 42-45 conv2 | 46 lin_b | 47 lin_w | 48 w_in |
//  49-52 wdown | 53-56 wup | 57-60 kpos | 61-65 mask | 66-70 nbr |
//  71-75 nsites | 76-79 parent
// ---------------------------------------------------------------------------
extern "C" void kernel_launch(void* const* d_in, const int* in_sizes, int n_in,
                              void* d_out, int out_size, void* d_ws, size_t ws_size,
                              hipStream_t stream) {
  (void)n_in; (void)out_size; (void)ws_size;
  static const int n[5] = {32, 64, 96, 128, 160};

  const float* feats = (const float*)d_in[0];
  const float *bn1g[5], *bn1b[5], *bn2g[4], *bn2b[4], *bndg[4], *bndb[4], *bnug[4], *bnub[4];
  for (int l = 0; l < 5; ++l) { bn1g[l] = (const float*)d_in[1 + 2*l];  bn1b[l] = (const float*)d_in[2 + 2*l]; }
  for (int l = 0; l < 4; ++l) { bn2g[l] = (const float*)d_in[11 + 2*l]; bn2b[l] = (const float*)d_in[12 + 2*l]; }
  for (int l = 0; l < 4; ++l) { bndg[l] = (const float*)d_in[19 + 2*l]; bndb[l] = (const float*)d_in[20 + 2*l]; }
  const float* bnfg = (const float*)d_in[27];
  const float* bnfb = (const float*)d_in[28];
  for (int l = 0; l < 4; ++l) { bnug[l] = (const float*)d_in[29 + 2*l]; bnub[l] = (const float*)d_in[30 + 2*l]; }
  const float* conv1[5]; for (int l = 0; l < 5; ++l) conv1[l] = (const float*)d_in[37 + l];
  const float* conv2[4]; for (int l = 0; l < 4; ++l) conv2[l] = (const float*)d_in[42 + l];
  const float* lin_b = (const float*)d_in[46];
  const float* lin_w = (const float*)d_in[47];
  const float* w_in  = (const float*)d_in[48];
  const float* wdown[4]; for (int l = 0; l < 4; ++l) wdown[l] = (const float*)d_in[49 + l];
  const float* wup[4];   for (int l = 0; l < 4; ++l) wup[l]   = (const float*)d_in[53 + l];
  const int* kp[4];  for (int l = 0; l < 4; ++l) kp[l]  = (const int*)d_in[57 + l];
  const unsigned char* mask[5]; for (int l = 0; l < 5; ++l) mask[l] = (const unsigned char*)d_in[61 + l];
  const int* nbr[5]; for (int l = 0; l < 5; ++l) nbr[l] = (const int*)d_in[66 + l];
  const int* par[4]; for (int l = 0; l < 4; ++l) par[l] = (const int*)d_in[76 + l];

  int N[5];
  for (int l = 0; l < 5; ++l) N[l] = in_sizes[66 + l] / 27;  // nbr leaf = N_l * 27

  // bump allocator over the workspace
  char* wp = (char*)d_ws;
  auto alloc = [&](size_t bytes) -> void* {
    void* p = (void*)wp;
    wp += (bytes + 255) & ~(size_t)255;
    return p;
  };
  float* hin[5]; float* x1[5]; unsigned short* b1[5]; unsigned short* wb1[5];
  float* df[4]; float* uf[4]; float* dec[4]; float* cat[4];
  unsigned short* b2[4]; unsigned short* wb2[4]; float* c2o[4];
  for (int l = 0; l < 5; ++l) {
    hin[l] = (float*)alloc((size_t)N[l] * n[l] * 4);
    x1[l]  = (float*)alloc((size_t)N[l] * n[l] * 4);
    b1[l]  = (unsigned short*)alloc((size_t)N[l] * n[l] * 2);
    wb1[l] = (unsigned short*)alloc((size_t)27 * n[l] * n[l] * 2);
  }
  for (int l = 0; l < 4; ++l) {
    df[l]  = (float*)alloc((size_t)N[l] * n[l] * 4);
    uf[l]  = (float*)alloc((size_t)N[l + 1] * n[l + 1] * 4);
    dec[l] = (float*)alloc((size_t)N[l] * n[l] * 4);
    cat[l] = (float*)alloc((size_t)N[l] * 2 * n[l] * 4);
    b2[l]  = (unsigned short*)alloc((size_t)N[l] * 2 * n[l] * 2);
    wb2[l] = (unsigned short*)alloc((size_t)27 * 2 * n[l] * n[l] * 2);
    c2o[l] = (float*)alloc((size_t)N[l] * n[l] * 4);
  }
  float* stats = (float*)alloc(2 * 512 * 4);
  float* ffin  = (float*)alloc((size_t)N[0] * 32 * 4);

  auto grid1 = [](size_t total, int blk) { return dim3((unsigned)((total + blk - 1) / blk)); };
  auto bnrun = [&](const float* x, int NN, int C, const float* g, const float* b,
                   float* of, unsigned short* ob) {
    bn_stats_kernel<<<dim3(C), dim3(256), 0, stream>>>(x, NN, C, stats);
    bn_apply_kernel<<<grid1((size_t)NN * C, 256), dim3(256), 0, stream>>>(x, stats, g, b, of, ob, NN, C);
  };

  // pack all 27-offset conv weights to bf16 [k][Cout][Cin]
  for (int l = 0; l < 5; ++l)
    wpack_kernel<<<grid1((size_t)27 * n[l] * n[l], 256), dim3(256), 0, stream>>>(
        conv1[l], wb1[l], 27, n[l], n[l]);
  for (int l = 0; l < 4; ++l)
    wpack_kernel<<<grid1((size_t)27 * 2 * n[l] * n[l], 256), dim3(256), 0, stream>>>(
        conv2[l], wb2[l], 27, 2 * n[l], n[l]);

  // initial 1->32 submanifold conv
  subconv_in_kernel<<<grid1((size_t)N[0] * 32, 256), dim3(256), 0, stream>>>(
      feats, w_in, nbr[0], mask[0], hin[0], N[0]);

  // encoder sweep
  for (int l = 0; l < 5; ++l) {
    bnrun(hin[l], N[l], n[l], bn1g[l], bn1b[l], nullptr, b1[l]);
    launch_subconv(b1[l], wb1[l], nbr[l], mask[l], x1[l], N[l], n[l], n[l], stream);
    if (l < 4) {
      bnrun(x1[l], N[l], n[l], bndg[l], bndb[l], df[l], nullptr);
      size_t zc = (size_t)N[l + 1] * n[l + 1];
      zero_kernel<<<grid1(zc, 256), dim3(256), 0, stream>>>(hin[l + 1], zc);
      downconv_kernel<<<grid1((size_t)N[l] * n[l + 1], 256), dim3(256), 0, stream>>>(
          df[l], par[l], kp[l], wdown[l], hin[l + 1], N[l], n[l], n[l + 1]);
    }
  }

  // decoder sweep
  for (int l = 3; l >= 0; --l) {
    const float* y = (l == 3) ? x1[4] : c2o[l + 1];
    bnrun(y, N[l + 1], n[l + 1], bnug[l], bnub[l], uf[l], nullptr);
    deconv_kernel<<<grid1((size_t)N[l] * n[l], 256), dim3(256), 0, stream>>>(
        uf[l], par[l], kp[l], wup[l], dec[l], N[l], n[l + 1], n[l]);
    concat_kernel<<<grid1((size_t)N[l] * 2 * n[l], 256), dim3(256), 0, stream>>>(
        x1[l], dec[l], cat[l], N[l], n[l], n[l]);
    bnrun(cat[l], N[l], 2 * n[l], bn2g[l], bn2b[l], nullptr, b2[l]);
    launch_subconv(b2[l], wb2[l], nbr[l], mask[l], c2o[l], N[l], 2 * n[l], n[l], stream);
  }

  // final BN+ReLU and linear head
  bnrun(c2o[0], N[0], 32, bnfg, bnfb, ffin, nullptr);
  linear_kernel<<<grid1((size_t)N[0], 256), dim3(256), 0, stream>>>(
      ffin, lin_w, lin_b, (float*)d_out, N[0]);
}
